// RecurrentPolicy_13606456394334
// MI455X (gfx1250) — compile-verified, matrix operand order checked by reading, not run on hardware
//
#include <hip/hip_runtime.h>
#include <hip/hip_bf16.h>

// ---------------------------------------------------------------------------
// Types for WMMA
// ---------------------------------------------------------------------------
typedef __bf16 bf16;
typedef __attribute__((ext_vector_type(16))) __bf16 v16bf;
typedef __attribute__((ext_vector_type(8)))  __bf16 v8bf;
typedef __attribute__((ext_vector_type(4)))  __bf16 v4bf;
typedef __attribute__((ext_vector_type(8)))  float  v8f;

#define BM 128
#define BN 128
#define BK 32
#define LDT 40   // padded bf16 row stride in LDS (80B -> conflict-free b128 reads)

// ---------------------------------------------------------------------------
// C[m][n] = sum_k A[m][k] * W[n][k] + bias[n]
// A: M x K row-major f32; W: N x K row-major f32 (i.e. B^T); C: M x N f32.
// fp32 -> bf16 on the fly (v_cvt_pk_bf16_f32), v_wmma_f32_16x16x32_bf16.
// Ping-pong LDS: one barrier per K-step, global loads overlapped with WMMA.
// M multiple of 128; N arbitrary (mask-hoisted edge); K multiple of 32.
// ---------------------------------------------------------------------------
__global__ __launch_bounds__(256) void gemm_bf16_wmma(
    const float* __restrict__ A, const float* __restrict__ W,
    const float* __restrict__ bias, float* __restrict__ C,
    int M, int N, int K)
{
  __shared__ bf16 sA[2][BM * LDT];
  __shared__ bf16 sB[2][BN * LDT];

  const int tid  = threadIdx.x;
  const int m0   = blockIdx.y * BM;
  const int n0   = blockIdx.x * BN;
  const int wid  = tid >> 5;
  const int lane = tid & 31;
  const int half = lane >> 4;     // 0: lanes 0-15, 1: lanes 16-31
  const int lr   = lane & 15;
  const int wm   = (wid & 3) * 32;   // 4 waves along M
  const int wn   = (wid >> 2) * 64;  // 2 waves along N

  // ---- hoisted per-thread tile-load descriptors (4 float4 slots each) ----
  const float* pA[4];
  const float* pB[4];
  float        mB[4];
  int          lds[4];
#pragma unroll
  for (int i = 0; i < 4; ++i) {
    int idx = tid + i * 256;          // 0..1023
    int row = idx >> 3;               // 0..127
    int kc  = (idx & 7) * 4;          // 0,4,...,28
    pA[i]  = A + (size_t)(m0 + row) * K + kc;
    int rB = n0 + row;
    mB[i]  = (rB < N) ? 1.0f : 0.0f;
    rB     = (rB < N) ? rB : (N - 1);
    pB[i]  = W + (size_t)rB * K + kc;
    lds[i] = row * LDT + kc;
  }

  v8f acc[2][4];
#pragma unroll
  for (int mi = 0; mi < 2; ++mi)
#pragma unroll
    for (int ni = 0; ni < 4; ++ni)
#pragma unroll
      for (int e = 0; e < 8; ++e) acc[mi][ni][e] = 0.0f;

  // ---- prologue: stage tile 0 ----
  float4 ra[4], rb[4];
#pragma unroll
  for (int i = 0; i < 4; ++i) { ra[i] = *(const float4*)pA[i]; rb[i] = *(const float4*)pB[i]; }
#pragma unroll
  for (int i = 0; i < 4; ++i) {
    v4bf pa; pa.x = (bf16)ra[i].x; pa.y = (bf16)ra[i].y; pa.z = (bf16)ra[i].z; pa.w = (bf16)ra[i].w;
    *(v4bf*)&sA[0][lds[i]] = pa;
    v4bf pb; pb.x = (bf16)(rb[i].x * mB[i]); pb.y = (bf16)(rb[i].y * mB[i]);
             pb.z = (bf16)(rb[i].z * mB[i]); pb.w = (bf16)(rb[i].w * mB[i]);
    *(v4bf*)&sB[0][lds[i]] = pb;
  }
  __syncthreads();

  const int nk = K / BK;
  for (int kt = 0; kt < nk; ++kt) {
    const int cur = kt & 1, nxt = cur ^ 1;
    const bool hasNext = (kt + 1) < nk;

    // issue next tile's global loads first (latency hidden by WMMAs below)
    if (hasNext) {
      int ko = (kt + 1) * BK;
#pragma unroll
      for (int i = 0; i < 4; ++i) {
        ra[i] = *(const float4*)(pA[i] + ko);
        rb[i] = *(const float4*)(pB[i] + ko);
      }
    }

    // fragment loads per ISA 16-bit A/B layout:
    // lanes 0-15: K chunks {0..7, 16..23}; lanes 16-31: {8..15, 24..31}
    v16bf aF[2], bF[4];
#pragma unroll
    for (int mi = 0; mi < 2; ++mi) {
      int r = wm + mi * 16 + lr;
      v8bf lo = *(const v8bf*)&sA[cur][r * LDT + half * 8];
      v8bf hi = *(const v8bf*)&sA[cur][r * LDT + 16 + half * 8];
      aF[mi] = __builtin_shufflevector(lo, hi, 0,1,2,3,4,5,6,7,8,9,10,11,12,13,14,15);
    }
#pragma unroll
    for (int ni = 0; ni < 4; ++ni) {
      int r = wn + ni * 16 + lr;
      v8bf lo = *(const v8bf*)&sB[cur][r * LDT + half * 8];
      v8bf hi = *(const v8bf*)&sB[cur][r * LDT + 16 + half * 8];
      bF[ni] = __builtin_shufflevector(lo, hi, 0,1,2,3,4,5,6,7,8,9,10,11,12,13,14,15);
    }

#pragma unroll
    for (int mi = 0; mi < 2; ++mi)
#pragma unroll
      for (int ni = 0; ni < 4; ++ni)
        acc[mi][ni] = __builtin_amdgcn_wmma_f32_16x16x32_bf16(
            false, aF[mi], false, bF[ni], (short)0, acc[mi][ni], false, false);

    // convert + park next tile into the other LDS buffer
    if (hasNext) {
#pragma unroll
      for (int i = 0; i < 4; ++i) {
        v4bf pa; pa.x = (bf16)ra[i].x; pa.y = (bf16)ra[i].y; pa.z = (bf16)ra[i].z; pa.w = (bf16)ra[i].w;
        *(v4bf*)&sA[nxt][lds[i]] = pa;
        v4bf pb; pb.x = (bf16)(rb[i].x * mB[i]); pb.y = (bf16)(rb[i].y * mB[i]);
                 pb.z = (bf16)(rb[i].z * mB[i]); pb.w = (bf16)(rb[i].w * mB[i]);
        *(v4bf*)&sB[nxt][lds[i]] = pb;
      }
    }
    __syncthreads();
  }

  // ---- store: C element (VGPR r, lane) -> M = r + 8*half, N = lr ----
#pragma unroll
  for (int ni = 0; ni < 4; ++ni) {
    int gn = n0 + wn + ni * 16 + lr;
    if (gn >= N) continue;
    float bv = bias ? bias[gn] : 0.0f;
#pragma unroll
    for (int mi = 0; mi < 2; ++mi) {
      int gmBase = m0 + wm + mi * 16 + half * 8;
#pragma unroll
      for (int r = 0; r < 8; ++r)
        C[(size_t)(gmBase + r) * N + gn] = acc[mi][ni][r] + bv;
    }
  }
}

// ---------------------------------------------------------------------------
// Persistent LSTM scan over T=512 steps (one layer).
// Grid: 48 WGs x 256 threads; WG w owns h-slice [w*16, w*16+16).
// Cross-WG step sync: monotonic device-scope atomic counter + spin (s_sleep).
// h state double-buffered in global hbuf[2][4][768]; the per-step h(t)
// broadcast into LDS uses global_load_async_to_lds_b128 (ASYNCcnt path).
// ---------------------------------------------------------------------------
#define SCAN_WGS 48
#define S_HH 16          // 768 / 48
#define TT 512
#define HH 768
#define HPAD 784         // 768+16: batch rows land in distinct LDS banks

__global__ __launch_bounds__(256) void lstm_scan(
    const float* __restrict__ xg,    // (B*T, 3072) row m = b*T+t
    const float* __restrict__ Whh,   // (3072, 768)
    float* __restrict__ h_seq,       // (B*T, 768)
    float* __restrict__ hbuf,        // 2 * 4 * 768
    unsigned* __restrict__ flag)     // monotonic arrival counter (pre-zeroed)
{
  __shared__ float hp[4 * HPAD];      // h_prev, all 4 batches
  __shared__ float sg[4][4][S_HH];    // [gate][b][hh]
  __shared__ float sc[4][S_HH];       // cell state owned by this WG

  const int tid = threadIdx.x;
  const int hh0 = blockIdx.x * S_HH;
  const int g   = tid >> 6;           // 0..3
  const int hh  = (tid >> 2) & 15;    // 0..15
  const int b   = tid & 3;            // 0..3
  const float* __restrict__ wrow = Whh + (size_t)(g * HH + hh0 + hh) * HH;
  const float* __restrict__ hb   = hp + b * HPAD;

  for (int i = tid; i < 4 * HPAD; i += 256) hp[i] = 0.0f;
  if (tid < 64) sc[tid >> 4][tid & 15] = 0.0f;
  __syncthreads();

  for (int t = 0; t < TT; ++t) {
    // recurrent matvec: one output per thread, W row streamed from L2
    float acc = 0.0f;
#pragma unroll 8
    for (int k = 0; k < HH; k += 4) {
      float4 w = *(const float4*)(wrow + k);
      acc += w.x * hb[k]     + w.y * hb[k + 1]
           + w.z * hb[k + 2] + w.w * hb[k + 3];
    }
    acc += xg[(size_t)(b * TT + t) * (4 * HH) + g * HH + hh0 + hh];
    sg[g][b][hh] = acc;
    __syncthreads();

    // pointwise LSTM update for this WG's slice
    if (tid < 64) {
      int pb = tid >> 4, ph = tid & 15;
      float iv = 1.0f / (1.0f + __expf(-sg[0][pb][ph]));
      float fv = 1.0f / (1.0f + __expf(-sg[1][pb][ph]));
      float gv = tanhf(sg[2][pb][ph]);
      float ov = 1.0f / (1.0f + __expf(-sg[3][pb][ph]));
      float c2 = fv * sc[pb][ph] + iv * gv;
      float h2 = ov * tanhf(c2);
      sc[pb][ph] = c2;
      h_seq[(size_t)(pb * TT + t) * HH + hh0 + ph] = h2;
      hbuf[(t & 1) * (4 * HH) + pb * HH + hh0 + ph] = h2;
    }

    // device-scope step barrier (release stores, acquire loads)
    __threadfence();
    __syncthreads();
    if (tid == 0) {
      __hip_atomic_fetch_add(flag, 1u, __ATOMIC_RELEASE, __HIP_MEMORY_SCOPE_AGENT);
      unsigned target = (unsigned)SCAN_WGS * (unsigned)(t + 1);
      while (__hip_atomic_load(flag, __ATOMIC_ACQUIRE, __HIP_MEMORY_SCOPE_AGENT) < target)
        __builtin_amdgcn_s_sleep(2);
    }
    __syncthreads();
    __threadfence();

    // pull full h(t) for the next step via async global->LDS DMA path.
    // 768 chunks of 16B; chunks never straddle the HPAD row padding.
    if (t < TT - 1) {
#pragma unroll
      for (int j = 0; j < 3; ++j) {
        int i  = tid + j * 256;            // 0..767 (float4 chunk id)
        int bb = i / 192;                  // batch (192 chunks per batch row)
        int kk = (i - bb * 192) * 4;       // element offset within row
        const float* src = hbuf + (t & 1) * (4 * HH) + i * 4;
        unsigned dst = (unsigned)(size_t)&hp[bb * HPAD + kk]; // LDS byte addr
        asm volatile("global_load_async_to_lds_b128 %0, %1, off"
                     :: "v"(dst), "v"(src) : "memory");
      }
#if __has_builtin(__builtin_amdgcn_s_wait_asynccnt)
      __builtin_amdgcn_s_wait_asynccnt(0);
#else
      asm volatile("s_wait_asynccnt 0x0" ::: "memory");
#endif
    }
    __syncthreads();
  }
}

// ---------------------------------------------------------------------------
// Host-side orchestration
// ---------------------------------------------------------------------------
extern "C" void kernel_launch(void* const* d_in, const int* in_sizes, int n_in,
                              void* d_out, int out_size, void* d_ws, size_t ws_size,
                              hipStream_t stream) {
  (void)in_sizes; (void)n_in; (void)out_size; (void)ws_size;
  const float* reps   = (const float*)d_in[0];  // (4,512,768)
  const float* W_ih   = (const float*)d_in[1];  // (2,3072,768)
  const float* W_hh   = (const float*)d_in[2];  // (2,3072,768)
  const float* bvec   = (const float*)d_in[3];  // (2,3072)
  const float* head_w = (const float*)d_in[4];  // (50257,768)
  const float* head_b = (const float*)d_in[5];  // (50257)
  float* out = (float*)d_out;                   // (4,512,50257)

  const int M = 4 * 512, D = 768, H = 768, G4 = 3072, V = 50257;

  float*    xg    = (float*)d_ws;                       // 2048*3072 f32 (25.2 MB)
  float*    hseq  = xg + (size_t)M * G4;                // 2048*768  f32 (6.3 MB)
  float*    hbuf  = hseq + (size_t)M * H;               // 2*4*768   f32
  unsigned* flags = (unsigned*)(hbuf + 2 * 4 * H);      // 2 counters

  hipMemsetAsync(flags, 0, 2 * sizeof(unsigned), stream);

  dim3 blk(256);
  // Layer 0: input projection + scan
  gemm_bf16_wmma<<<dim3(G4 / BN, M / BM), blk, 0, stream>>>(
      reps, W_ih, bvec, xg, M, G4, D);
  lstm_scan<<<dim3(SCAN_WGS), blk, 0, stream>>>(xg, W_hh, hseq, hbuf, flags + 0);

  // Layer 1: input projection + scan (reuse xg / hseq sequentially)
  gemm_bf16_wmma<<<dim3(G4 / BN, M / BM), blk, 0, stream>>>(
      hseq, W_ih + (size_t)G4 * D, bvec + G4, xg, M, G4, H);
  lstm_scan<<<dim3(SCAN_WGS), blk, 0, stream>>>(
      xg, W_hh + (size_t)G4 * H, hseq, hbuf, flags + 1);

  // Head: logits = hseq @ head_w^T + head_b
  gemm_bf16_wmma<<<dim3((V + BN - 1) / BN, M / BM), blk, 0, stream>>>(
      hseq, head_w, head_b, out, M, V, H);
}
